// QNNLinear_12180527251554
// MI455X (gfx1250) — matrix operand order repaired and checked by reading, not compiled, not run on hardware
//
#include <hip/hip_runtime.h>

#define NQ 12
#define NL 6
#define BATCH 256
#define NSTATES 4096            // 2^NQ
#define TPB 256
#define APT (NSTATES / TPB)     // 16 amplitudes per thread
#define PPT (NSTATES / 2 / TPB) // 8 gate pairs per thread
#define CPT (NSTATES / 4 / TPB) // 4 cnot pairs per thread

typedef __attribute__((ext_vector_type(16))) _Float16 v16h;
typedef __attribute__((ext_vector_type(8)))  float    v8f;

// ---------------------------------------------------------------------------
// Generic 1-qubit complex 2x2 gate on the LDS state (interleaved float2).
// Ownership: thread tid owns all addresses with (addr & 0xFF) == tid, since
// pairs are enumerated as p = tid + k*TPB. A gate on bit >= 8 touches only
// owned addresses, so consecutive local gates need no barrier (sync=false).
// ---------------------------------------------------------------------------
__device__ __forceinline__ void apply_1q(float2* st, int bit, bool sync,
                                         float m00r, float m00i,
                                         float m01r, float m01i,
                                         float m10r, float m10i,
                                         float m11r, float m11i, int tid)
{
    if (sync) __syncthreads();
    const unsigned mask = (1u << bit) - 1u;
#pragma unroll
    for (int k = 0; k < PPT; ++k) {
        unsigned p  = (unsigned)tid + (unsigned)k * TPB;       // 0..2047
        unsigned i0 = ((p & ~mask) << 1) | (p & mask);
        unsigned i1 = i0 | (1u << bit);
        float2 a0 = st[i0];                                    // ds_load_b64
        float2 a1 = st[i1];
        float2 r0, r1;
        r0.x = m00r * a0.x - m00i * a0.y + m01r * a1.x - m01i * a1.y;
        r0.y = m00r * a0.y + m00i * a0.x + m01r * a1.y + m01i * a1.x;
        r1.x = m10r * a0.x - m10i * a0.y + m11r * a1.x - m11i * a1.y;
        r1.y = m10r * a0.y + m10i * a0.x + m11r * a1.y + m11i * a1.x;
        st[i0] = r0;                                           // ds_store_b64
        st[i1] = r1;
    }
}

// ---------------------------------------------------------------------------
// Kernel 1: full state-vector simulation, one workgroup per batch element.
// State lives entirely in LDS (4096 x float2 = 32 KB). Emits q = <Z_i>.
// ---------------------------------------------------------------------------
__global__ __launch_bounds__(TPB) void qnn_state_kernel(
    const float* __restrict__ s,        // (BATCH, NQ)
    const float* __restrict__ qp,       // (NL, 2*NQ)
    float* __restrict__ qout)           // (BATCH, NQ) scratch
{
    __shared__ float2 st[NSTATES];
    __shared__ float  zacc[NQ];

    const int b   = blockIdx.x;
    const int tid = threadIdx.x;

    // |0...0>   (writes only owned addresses -> no barrier before local gates)
#pragma unroll
    for (int k = 0; k < APT; ++k) {
        int idx = tid + k * TPB;
        st[idx] = make_float2((idx == 0) ? 1.0f : 0.0f, 0.0f);
    }
    if (tid < NQ) zacc[tid] = 0.0f;

    // Data-encoding layer: RX(s[b,i]*pi) on wire i (bit NQ-1-i).
    // Wires 0..3 -> bits 11..8: thread-local, no barrier between them.
#pragma unroll 1
    for (int i = 0; i < NQ; ++i) {
        float half = s[b * NQ + i] * 1.57079632679489662f;  // theta/2 = s*pi/2
        float c, sn;
        __sincosf(half, &sn, &c);
        // RX = [[c, -i s], [-i s, c]]
        apply_1q(st, NQ - 1 - i, /*sync=*/(i >= 4),
                 c, 0.0f,  0.0f, -sn,
                 0.0f, -sn,  c, 0.0f, tid);
    }

    // Variational layers
#pragma unroll 1
    for (int l = 0; l < NL; ++l) {
        // CNOT chain: ctrl=i (bit bc), tgt=i+1 (bit bc-1).
        // i=0..2 use bits >= 8 (local); sync needed entering i==0 (prev gate
        // was cross-thread) and for every cross-thread CNOT (i>=3).
#pragma unroll 1
        for (int i = 0; i < NQ - 1; ++i) {
            int bc = NQ - 1 - i, bt = bc - 1;
            if (i == 0 || i >= 3) __syncthreads();
            unsigned mask0 = (1u << bt) - 1u;   // bt < bc
            unsigned mask1 = (1u << bc) - 1u;
#pragma unroll
            for (int k = 0; k < CPT; ++k) {
                unsigned p  = (unsigned)tid + (unsigned)k * TPB;  // 0..1023
                unsigned x  = ((p & ~mask0) << 1) | (p & mask0);
                unsigned y  = ((x & ~mask1) << 1) | (x & mask1);
                unsigned ia = y | (1u << bc);          // ctrl=1, tgt=0
                unsigned ib = ia | (1u << bt);         // ctrl=1, tgt=1
                float2 t = st[ia];
                st[ia] = st[ib];
                st[ib] = t;
            }
        }
        // Fused M = RZ(b) * RX(a) per wire. Wires 0..3 local; sync entering
        // wire 0 (prev CNOT was cross) and every cross wire (i>=4).
#pragma unroll 1
        for (int i = 0; i < NQ; ++i) {
            float ah = qp[l * 2 * NQ + 2 * i]     * 0.5f;
            float bh = qp[l * 2 * NQ + 2 * i + 1] * 0.5f;
            float c, sn, zr, zi;
            __sincosf(ah, &sn, &c);
            __sincosf(bh, &zi, &zr);
            // e = zr - i*zi ; M = RZ*RX
            apply_1q(st, NQ - 1 - i, /*sync=*/(i == 0 || i >= 4),
                     zr * c,  -zi * c,
                     -sn * zi, -sn * zr,
                      sn * zi, -sn * zr,
                     zr * c,   zi * c, tid);
        }
    }

    __syncthreads();   // last gate was cross-thread; make all writes visible

    // <Z_i> = sum_amp prob * (+1 if bit==0 else -1); reads only owned addrs.
    float zp[NQ];
#pragma unroll
    for (int i = 0; i < NQ; ++i) zp[i] = 0.0f;
#pragma unroll
    for (int k = 0; k < APT; ++k) {
        int    idx = tid + k * TPB;
        float2 a   = st[idx];
        float  pr  = a.x * a.x + a.y * a.y;
#pragma unroll
        for (int i = 0; i < NQ; ++i) {
            unsigned bit = (unsigned)(idx >> (NQ - 1 - i)) & 1u;
            zp[i] += __int_as_float(__float_as_int(pr) ^ (bit << 31));
        }
    }
#pragma unroll
    for (int i = 0; i < NQ; ++i) atomicAdd(&zacc[i], zp[i]);  // ds_add_f32
    __syncthreads();
    if (tid < NQ) qout[b * NQ + tid] = zacc[tid];
}

// ---------------------------------------------------------------------------
// Lane XOR-shuffle via ds_swizzle (group-of-32 pattern: offset =
// xor_mask<<10 | or_mask<<5 | and_mask). Avoids bpermute+mask codegen.
// ---------------------------------------------------------------------------
template <int XORMASK>
__device__ __forceinline__ float swz_xor_add(float v)
{
    int sw = __builtin_amdgcn_ds_swizzle(__float_as_int(v),
                                         (XORMASK << 10) | 0x1f);
    return v + __int_as_float(sw);
}

// ---------------------------------------------------------------------------
// Kernel 2: head  out[m] = sum_j relu( sum_k q[m,k] * W[k,j] )
// One wave per 16-row tile; single v_wmma_f32_16x16x32_f16 per wave
// (K padded 12->32, N padded 12->16 with zeros).
// ---------------------------------------------------------------------------
__global__ __launch_bounds__(32) void qnn_head_kernel(
    const float* __restrict__ q,   // (BATCH, NQ)
    const float* __restrict__ W,   // (NQ, NQ)
    float* __restrict__ out)       // (BATCH,)
{
    const int lane = threadIdx.x;        // 0..31
    const int tile = blockIdx.x;         // 16 tiles of 16 rows

    // A fragment: 16x32 f16. Lanes 0-15: M=lane, halves 0..7 = K 0..7,
    // halves 8..15 = K 16..23. Lanes 16-31: M=lane-16, halves 0..7 = K 8..15.
    v16h a = {};
    {
        int m     = tile * 16 + (lane & 15);
        int kbase = (lane < 16) ? 0 : 8;
#pragma unroll
        for (int h = 0; h < 8; ++h) {
            int K = kbase + h;
            if (K < NQ) a[h] = (_Float16)q[m * NQ + K];
        }
        // halves 8..15 correspond to K>=16 -> stay zero
    }

    // B fragment: 32x16 f16. Lanes 0-15: N=lane, halves 0..7 = K 0..7,
    // halves 8..15 = K 16..23. Lanes 16-31: N=lane-16, halves 0..7 = K 8..15.
    v16h bf = {};
    {
        int n = lane & 15;
        int kbase = (lane < 16) ? 0 : 8;
        if (n < NQ) {
#pragma unroll
            for (int h = 0; h < 8; ++h) {
                int K = kbase + h;
                if (K < NQ) bf[h] = (_Float16)W[K * NQ + n];
            }
        }
    }

    v8f c = {};
    c = __builtin_amdgcn_wmma_f32_16x16x32_f16(
            /*neg_a=*/false, a, /*neg_b=*/false, bf,
            /*c_mod=*/(short)0, c, /*reuse_a=*/false, /*reuse_b=*/false);

    // C layout: VGPR r -> row (r or r+8), lane%16 -> column n.
    int n = lane & 15;
#pragma unroll
    for (int r = 0; r < 8; ++r)
        c[r] = (n < NQ) ? fmaxf(c[r], 0.0f) : 0.0f;   // relu + mask pad cols

    // Butterfly sum over the 16 lanes of each half (columns 0..15).
#pragma unroll
    for (int r = 0; r < 8; ++r) {
        float v = c[r];
        v = swz_xor_add<1>(v);
        v = swz_xor_add<2>(v);
        v = swz_xor_add<4>(v);
        v = swz_xor_add<8>(v);
        c[r] = v;
    }
    if ((lane & 15) == 0) {
        int base = tile * 16 + ((lane >> 4) << 3);    // lane 0 -> rows 0..7, lane 16 -> 8..15
#pragma unroll
        for (int r = 0; r < 8; ++r) out[base + r] = c[r];
    }
}

// ---------------------------------------------------------------------------
extern "C" void kernel_launch(void* const* d_in, const int* in_sizes, int n_in,
                              void* d_out, int out_size, void* d_ws, size_t ws_size,
                              hipStream_t stream)
{
    const float* s  = (const float*)d_in[0];   // (256, 12)
    const float* qp = (const float*)d_in[1];   // (6, 24)
    const float* W  = (const float*)d_in[2];   // (12, 12)
    float* out      = (float*)d_out;           // (256,)
    float* q_ws     = (float*)d_ws;            // (256, 12) scratch

    qnn_state_kernel<<<BATCH, TPB, 0, stream>>>(s, qp, q_ws);
    qnn_head_kernel<<<BATCH / 16, 32, 0, stream>>>(q_ws, W, out);
}
